// ONLSTM_18485539242474
// MI455X (gfx1250) — matrix operand order repaired and checked by reading, not compiled, not run on hardware
//
#include <hip/hip_runtime.h>

// ---------------------------------------------------------------------------
// ON-LSTM fused kernel for MI455X (gfx1250, wave32, WMMA bf16 16x16x32).
//
//  * 9 GEMMs as split-precision bf16 WMMA (hi/lo split, 3 wmma per tile-step)
//  * fp32 -> bf16 hi/lo split done ONCE at staging (cooperatively); per-wave
//    A-fragment build is 2x ds_load_b128 + a register shuffle (no cvt VALU)
//  * BM=64 rows per workgroup, 16 waves; wave = 1 n-tile x 4 M-tiles so each
//    B-fragment pair (2 KB from L2) feeds 12 WMMAs -> halves L2 traffic vs
//    BM=32 (B-fragment stream is the dominant byte stream after HBM)
//  * h_new@u_f fused via LDS round-trip (no global sync)
//  * cumax (cumsum of softmax over H) via wave32 shuffle reduce + scan
//  * activations use hardware v_exp_f32 / v_rcp_f32 paths (libm tanhf/expf
//    expansions were the largest remaining VALU stream)
//  * weights pre-swizzled once into WMMA B-fragment layout in d_ws (2 MB)
// ---------------------------------------------------------------------------

typedef __attribute__((ext_vector_type(16))) __bf16 v16bf;
typedef __attribute__((ext_vector_type(8)))  __bf16 v8bf;
typedef __attribute__((ext_vector_type(8)))  float  v8f;

#define NB  65536     // batch
#define DH  256       // D == H == 256
#define BM  64        // batch rows per workgroup
#define STB 264       // bf16 LDS row stride (elements); 528 B -> conflict-free
#define STF 260       // fp32 LDS row stride (elements); 1040 B -> conflict-free

// LDS arena: four bf16 planes (64 x 264 x 2 B = 33792 B each), aliased later.
#define OFF_XH  0                      // x hi   / later h_new hi / later f_t
#define OFF_XL  33792                  // x lo   / later h_new lo
#define OFF_HH  67584                  // h hi   / later pre fp32 / later i_t
#define OFF_HL  101376                 // h lo
#define SMEM_SZ 135168                 // 132 KB
// fp32 aliases (64*260*4 = 66560 B each, fit in two adjacent bf16 planes):
//   pre @ OFF_HH (h dead after phase 1b)
//   f_t @ OFF_XH (h_new planes dead after phase 2)
//   i_t @ OFF_HH (overwrites pre in place; row read into regs before write)

static __device__ __forceinline__ v8f wmma_bf16(v16bf a, v16bf b, v8f c) {
  // D = A(16x32) * B(32x16) + C, fp32 accumulate
  return __builtin_amdgcn_wmma_f32_16x16x32_bf16(
      /*neg_a=*/false, a, /*neg_b=*/false, b,
      /*c_mod=*/(short)0, c, /*reuse_a=*/false, /*reuse_b=*/false);
}

// fast transcendentals on the hardware special-function pipe (~1 ulp)
static __device__ __forceinline__ float rcp_fast(float x) {
  return __builtin_amdgcn_rcpf(x);
}
static __device__ __forceinline__ float exp_fast(float x) {
  return __expf(x);                       // v_exp_f32 (x * log2e)
}
static __device__ __forceinline__ float sigmoid_fast(float x) {
  return rcp_fast(1.0f + exp_fast(-x));
}
static __device__ __forceinline__ float tanh_fast(float x) {
  // 1 - 2/(e^{2x}+1); exact limits at +-inf via IEEE semantics
  return 1.0f - 2.0f * rcp_fast(exp_fast(2.0f * x) + 1.0f);
}

// A-fragment (16x32 bf16 tile) load from a bf16 LDS plane stored row-major.
// p = plane + row*STB + (lane>>4)*8 + ks*32.  Element e of the fragment maps
// to K offset (e<8 ? e : e+8), i.e. bytes [0,16) and [32,48) from p.
static __device__ __forceinline__ v16bf load_afrag(const __bf16* __restrict__ p) {
  v8bf lo8 = *(const v8bf*)(p);        // K + 0..7   (ds_load_b128)
  v8bf hi8 = *(const v8bf*)(p + 16);   // K + 16..23 (ds_load_b128)
  return __builtin_shufflevector(lo8, hi8, 0, 1, 2, 3, 4, 5, 6, 7,
                                 8, 9, 10, 11, 12, 13, 14, 15);
}

// split one fp32 value into bf16 hi + lo
static __device__ __forceinline__ void split1(float v, __bf16* hb, __bf16* lb) {
  __bf16 h = (__bf16)v;
  *hb = h;
  *lb = (__bf16)(v - (float)h);
}

// ---------------------------------------------------------------------------
// Prep: swizzle the 8 weight matrices (256x256 fp32, row = K, col = N) into
// per-lane WMMA B-fragments, bf16 hi + lo.  Layout (as v16bf units):
//   frag[(((mat*2 + prec)*16 + ntile)*8 + kstep)*32 + lane]
// One wave per (mat, ntile, kstep): 8*16*8 = 1024 blocks of 32 threads.
// ---------------------------------------------------------------------------
__global__ __launch_bounds__(32) void onlstm_prep_weights(
    const float* __restrict__ w_f, const float* __restrict__ w_i,
    const float* __restrict__ w_o, const float* __restrict__ w_c,
    const float* __restrict__ u_f, const float* __restrict__ u_i,
    const float* __restrict__ u_o, const float* __restrict__ u_c,
    __bf16* __restrict__ wsB) {
  const float* Ws[8] = {w_f, w_i, w_o, w_c, u_f, u_i, u_o, u_c};
  const int blk  = blockIdx.x;
  const int mat  = blk >> 7;
  const int nt   = (blk >> 3) & 15;
  const int ks   = blk & 7;
  const int lane = threadIdx.x;

  const float* W = Ws[mat];
  const int n  = nt * 16 + (lane & 15);
  const int kb = ks * 32 + ((lane >> 4) << 3);   // B layout: K = (e<8 ? e : e+8) + 8*(lane>=16)

  __bf16* oh = wsB + (size_t)(((((mat * 2 + 0) * 16 + nt) * 8 + ks) * 32) + lane) * 16;
  __bf16* ol = wsB + (size_t)(((((mat * 2 + 1) * 16 + nt) * 8 + ks) * 32) + lane) * 16;
#pragma unroll
  for (int e = 0; e < 16; ++e) {
    float v = W[(size_t)(kb + (e < 8 ? e : e + 8)) * DH + n];
    __bf16 hb, lb;
    split1(v, &hb, &lb);
    oh[e] = hb;
    ol[e] = lb;
  }
}

// ---------------------------------------------------------------------------
// Main fused kernel. 512 threads = 16 waves; wave wv owns the 16 output cols
// [wv*16, wv*16+16) across all 64 rows of the block (4 M-tiles, 1 N-tile).
// ---------------------------------------------------------------------------
__global__ __launch_bounds__(512) void onlstm_main(
    const float* __restrict__ x, const float* __restrict__ h,
    const float* __restrict__ c_prev,
    const float* __restrict__ b_f, const float* __restrict__ b_i,
    const float* __restrict__ b_o, const float* __restrict__ b_c,
    const __bf16* __restrict__ wsB,
    float* __restrict__ out_h, float* __restrict__ out_c) {
  __shared__ __align__(16) unsigned char smem[SMEM_SZ];
  __bf16* Xh = (__bf16*)(smem + OFF_XH);   // x hi,  later h_new hi
  __bf16* Xl = (__bf16*)(smem + OFF_XL);   // x lo,  later h_new lo
  __bf16* Hh = (__bf16*)(smem + OFF_HH);   // h hi
  __bf16* Hl = (__bf16*)(smem + OFF_HL);   // h lo
  float*  Pre = (float*)(smem + OFF_HH);   // phase 2: pre (aliases Hh/Hl)
  float*  Ft  = (float*)(smem + OFF_XH);   // phase 3: f_t (aliases Xh/Xl)
  float*  It  = (float*)(smem + OFF_HH);   // phase 3: i_t (overwrites pre)

  const int  tid  = threadIdx.x;
  const int  lane = tid & 31;
  const int  wv   = tid >> 5;              // 0..15 = n-tile
  const long row0 = (long)blockIdx.x * BM;

  const int m15  = lane & 15;
  const int hi16 = lane >> 4;              // 0 or 1

  // ---- stage x and h tiles, splitting fp32 -> bf16 hi/lo ONCE --------------
#pragma unroll
  for (int i = tid; i < BM * 64; i += 512) {
    const int r = i >> 6;
    const int c = (i & 63) * 4;
    float4 xv = ((const float4*)(x + (row0 + r) * DH))[i & 63];
    float4 hv = ((const float4*)(h + (row0 + r) * DH))[i & 63];
    float xa[4] = {xv.x, xv.y, xv.z, xv.w};
    float ha[4] = {hv.x, hv.y, hv.z, hv.w};
    __bf16* pxh = Xh + r * STB + c;
    __bf16* pxl = Xl + r * STB + c;
    __bf16* phh = Hh + r * STB + c;
    __bf16* phl = Hl + r * STB + c;
#pragma unroll
    for (int j = 0; j < 4; ++j) {
      split1(xa[j], &pxh[j], &pxl[j]);
      split1(ha[j], &phh[j], &phl[j]);
    }
  }
  __syncthreads();

  const v16bf* B = (const v16bf*)wsB;
  // A-fragment per-lane base offset within a plane (add mt*16*STB + ks*32)
  const int aoff = m15 * STB + hi16 * 8;

  v8f acc[4][4];                           // [gate][mtile]
#pragma unroll
  for (int g = 0; g < 4; ++g)
#pragma unroll
    for (int m = 0; m < 4; ++m)
#pragma unroll
      for (int v = 0; v < 8; ++v) acc[g][m][v] = 0.0f;

  // ---- phase 1a: acc[g] = x @ w_g  (g: 0=f 1=i 2=o 3=c) --------------------
  for (int ks = 0; ks < 8; ++ks) {
    v16bf ah[4], al[4];
#pragma unroll
    for (int m = 0; m < 4; ++m) {
      ah[m] = load_afrag(Xh + aoff + m * 16 * STB + ks * 32);
      al[m] = load_afrag(Xl + aoff + m * 16 * STB + ks * 32);
    }
#pragma unroll
    for (int g = 0; g < 4; ++g) {
      v16bf bh = B[((((g * 2 + 0) * 16 + wv) * 8 + ks) * 32) + lane];
      v16bf bl = B[((((g * 2 + 1) * 16 + wv) * 8 + ks) * 32) + lane];
#pragma unroll
      for (int m = 0; m < 4; ++m) {
        acc[g][m] = wmma_bf16(ah[m], bh, acc[g][m]);
        acc[g][m] = wmma_bf16(al[m], bh, acc[g][m]);
        acc[g][m] = wmma_bf16(ah[m], bl, acc[g][m]);
      }
    }
  }
  // keep xf = x @ w_f for the master-gate pre-activation
  v8f xf[4] = {acc[0][0], acc[0][1], acc[0][2], acc[0][3]};

  // ---- phase 1b: acc[g] += h @ u_g ----------------------------------------
  for (int ks = 0; ks < 8; ++ks) {
    v16bf ah[4], al[4];
#pragma unroll
    for (int m = 0; m < 4; ++m) {
      ah[m] = load_afrag(Hh + aoff + m * 16 * STB + ks * 32);
      al[m] = load_afrag(Hl + aoff + m * 16 * STB + ks * 32);
    }
#pragma unroll
    for (int g = 0; g < 4; ++g) {
      const int mm = g + 4;                // u_f..u_c
      v16bf bh = B[((((mm * 2 + 0) * 16 + wv) * 8 + ks) * 32) + lane];
      v16bf bl = B[((((mm * 2 + 1) * 16 + wv) * 8 + ks) * 32) + lane];
#pragma unroll
      for (int m = 0; m < 4; ++m) {
        acc[g][m] = wmma_bf16(ah[m], bh, acc[g][m]);
        acc[g][m] = wmma_bf16(al[m], bh, acc[g][m]);
        acc[g][m] = wmma_bf16(ah[m], bl, acc[g][m]);
      }
    }
  }

  // ---- gate activations in D layout; write h_new to global -----------------
  float fg[4][8], ig[4][8], cg[4][8], hn[4][8];
  {
    const int col = wv * 16 + m15;
    const float bfv = b_f[col], biv = b_i[col], bov = b_o[col], bcv = b_c[col];
#pragma unroll
    for (int m = 0; m < 4; ++m)
#pragma unroll
      for (int v = 0; v < 8; ++v) {
        const int  M = m * 16 + v + hi16 * 8;
        const float fv = sigmoid_fast(acc[0][m][v] + bfv);
        const float iv = sigmoid_fast(acc[1][m][v] + biv);
        const float ov = sigmoid_fast(acc[2][m][v] + bov);
        const float cv = tanh_fast(acc[3][m][v] + bcv);
        const float hv = ov * tanh_fast(cv);
        fg[m][v] = fv; ig[m][v] = iv; cg[m][v] = cv; hn[m][v] = hv;
        out_h[(row0 + M) * DH + col] = hv;
      }
  }
  __syncthreads();                          // everyone done reading Xh/Xl/Hh/Hl

  // ---- h_new -> bf16 hi/lo planes (reuse Xh/Xl) ----------------------------
  {
    const int col = wv * 16 + m15;
#pragma unroll
    for (int m = 0; m < 4; ++m)
#pragma unroll
      for (int v = 0; v < 8; ++v) {
        const int M = m * 16 + v + hi16 * 8;
        split1(hn[m][v], &Xh[M * STB + col], &Xl[M * STB + col]);
      }
  }
  __syncthreads();

  // ---- phase 2: pre = xf + h_new @ u_f ------------------------------------
  v8f pre2[4] = {xf[0], xf[1], xf[2], xf[3]};
  for (int ks = 0; ks < 8; ++ks) {
    v16bf ah[4], al[4];
#pragma unroll
    for (int m = 0; m < 4; ++m) {
      ah[m] = load_afrag(Xh + aoff + m * 16 * STB + ks * 32);
      al[m] = load_afrag(Xl + aoff + m * 16 * STB + ks * 32);
    }
    v16bf bh = B[((((4 * 2 + 0) * 16 + wv) * 8 + ks) * 32) + lane];  // u_f hi
    v16bf bl = B[((((4 * 2 + 1) * 16 + wv) * 8 + ks) * 32) + lane];  // u_f lo
#pragma unroll
    for (int m = 0; m < 4; ++m) {
      pre2[m] = wmma_bf16(ah[m], bh, pre2[m]);
      pre2[m] = wmma_bf16(al[m], bh, pre2[m]);
      pre2[m] = wmma_bf16(ah[m], bl, pre2[m]);
    }
  }
  // pre -> fp32 region aliasing Hh/Hl (h is dead; barrier above covers reads)
  {
    const int col = wv * 16 + m15;
#pragma unroll
    for (int m = 0; m < 4; ++m)
#pragma unroll
      for (int v = 0; v < 8; ++v)
        Pre[(m * 16 + v + hi16 * 8) * STF + col] = pre2[m][v];
  }
  __syncthreads();

  // ---- phase 3a: row-wise cumax twice (b_f -> f_t, b_i -> i_t) -------------
  // 16 waves x 4 rows each; lane owns 8 contiguous cols of the row.
  {
    const int c0 = lane * 8;
    float bf8[8], bi8[8];
#pragma unroll
    for (int j = 0; j < 8; ++j) { bf8[j] = b_f[c0 + j]; bi8[j] = b_i[c0 + j]; }

#pragma unroll
    for (int rr = 0; rr < 4; ++rr) {
      const int r = wv * 4 + rr;
      float pv[8];
#pragma unroll
      for (int j = 0; j < 8; ++j) pv[j] = Pre[r * STF + c0 + j];

      float ftv[8], itv[8];
#pragma unroll
      for (int variant = 0; variant < 2; ++variant) {
        float vb[8];
        float mx = -3.402823466e38f;
#pragma unroll
        for (int j = 0; j < 8; ++j) {
          vb[j] = pv[j] + (variant ? bi8[j] : bf8[j]);
          mx = fmaxf(mx, vb[j]);
        }
#pragma unroll
        for (int off = 16; off >= 1; off >>= 1)
          mx = fmaxf(mx, __shfl_xor(mx, off, 32));
        float run = 0.0f, pref[8];
#pragma unroll
        for (int j = 0; j < 8; ++j) {
          float e = exp_fast(vb[j] - mx);
          run += e;
          pref[j] = run;
        }
        float scan = run;                    // wave32 inclusive scan
#pragma unroll
        for (int off = 1; off < 32; off <<= 1) {
          float tsc = __shfl_up(scan, off, 32);
          if (lane >= off) scan += tsc;
        }
        const float excl  = scan - run;
        const float total = __shfl(scan, 31, 32);
        const float inv   = rcp_fast(total);
#pragma unroll
        for (int j = 0; j < 8; ++j) {
          const float cm = (excl + pref[j]) * inv;   // cumsum(softmax)
          if (variant == 0) ftv[j] = cm;
          else              itv[j] = 1.0f - cm;
        }
      }
      // stash f_t / i_t (this wave owns row r; i_t overwrites pre in place)
#pragma unroll
      for (int j = 0; j < 8; ++j) {
        Ft[r * STF + c0 + j] = ftv[j];
        It[r * STF + c0 + j] = itv[j];
      }
    }
  }
  __syncthreads();

  // ---- phase 3b: combine in D layout, write c_t ----------------------------
  {
    const int col = wv * 16 + m15;
#pragma unroll
    for (int m = 0; m < 4; ++m)
#pragma unroll
      for (int v = 0; v < 8; ++v) {
        const int   M    = m * 16 + v + hi16 * 8;
        const long  rowg = row0 + M;
        const float ft   = Ft[M * STF + col];
        const float it   = It[M * STF + col];
        const float f    = fg[m][v];
        const float i    = ig[m][v];
        const float ch   = cg[m][v];
        const float cp   = c_prev[rowg * DH + col];
        const float fth  = ft * (f * it + 1.0f - it);
        const float ith  = it * (i * ft + 1.0f - ft);
        out_c[rowg * DH + col] = fth * cp + ith + ch;
      }
  }
}

// ---------------------------------------------------------------------------
extern "C" void kernel_launch(void* const* d_in, const int* in_sizes, int n_in,
                              void* d_out, int out_size, void* d_ws, size_t ws_size,
                              hipStream_t stream) {
  const float* x      = (const float*)d_in[0];
  const float* h      = (const float*)d_in[1];
  const float* c_prev = (const float*)d_in[2];
  const float* w_f    = (const float*)d_in[3];
  const float* w_i    = (const float*)d_in[4];
  const float* w_o    = (const float*)d_in[5];
  const float* w_c    = (const float*)d_in[6];
  const float* u_f    = (const float*)d_in[7];
  const float* u_i    = (const float*)d_in[8];
  const float* u_o    = (const float*)d_in[9];
  const float* u_c    = (const float*)d_in[10];
  const float* b_f    = (const float*)d_in[11];
  const float* b_i    = (const float*)d_in[12];
  const float* b_o    = (const float*)d_in[13];
  const float* b_c    = (const float*)d_in[14];

  float* out_h = (float*)d_out;
  float* out_c = out_h + (size_t)NB * DH;
  __bf16* wsB  = (__bf16*)d_ws;              // needs 2 MB of scratch

  // swizzle the 8 weight matrices into bf16 hi/lo WMMA B-fragments
  onlstm_prep_weights<<<8 * 16 * 8, 32, 0, stream>>>(
      w_f, w_i, w_o, w_c, u_f, u_i, u_o, u_c, wsB);

  // fused ON-LSTM: 1024 blocks x 512 threads (16 waves), 64 batch rows each
  onlstm_main<<<NB / BM, 512, 0, stream>>>(
      x, h, c_prev, b_f, b_i, b_o, b_c, wsB, out_h, out_c);
}
